// Attention_72310069395589
// MI455X (gfx1250) — compile-verified, matrix operand order checked by reading, not run on hardware
//
#include <hip/hip_runtime.h>
#include <hip/hip_bf16.h>
#include <math.h>

typedef __attribute__((ext_vector_type(16))) _Float16 v16h;
typedef __attribute__((ext_vector_type(8)))  _Float16 v8h;
typedef __attribute__((ext_vector_type(8)))  float    v8f;

#define DIMD  1024
#define SEQ   2048
#define BATCH 4

// ---------------------------------------------------------------------------
// f32 -> f16 conversion
// ---------------------------------------------------------------------------
__global__ void cvt_f32_f16(const float* __restrict__ in, _Float16* __restrict__ out, int n) {
    int i = blockIdx.x * blockDim.x + threadIdx.x;
    if (i < n) out[i] = (_Float16)in[i];
}

// ---------------------------------------------------------------------------
// WMMA fragment gather from a per-lane pointer that already encodes
// (row0 + lane%16)*ld + 8*(lane/16).  Layout per ISA 16-bit A/B fragment:
// elements 0..7  = K [8*half, +8), elements 8..15 = K [16+8*half, +8).
// ---------------------------------------------------------------------------
__device__ __forceinline__ v16h make_frag(const _Float16* __restrict__ p) {
    v8h lo = *(const v8h*)(p);
    v8h hi = *(const v8h*)(p + 16);
    v16h out;
#pragma unroll
    for (int i = 0; i < 8; ++i) { out[i] = lo[i]; out[i + 8] = hi[i]; }
    return out;
}

// ---------------------------------------------------------------------------
// Generic WMMA GEMM:  C[M x N] = A[M x K] * B[N x K]^T   (A, B row-major f16)
// WG = 128 threads = 4 waves (2x2), WG tile 128x128, wave tile 64x64
// (4x4 WMMA tiles -> 16 WMMAs per k-step from 8 fragment loads).
// grid.x = N/128, grid.y = M/128. Dims must divide evenly (they do here).
// ---------------------------------------------------------------------------
template <bool F16OUT>
__global__ __launch_bounds__(128)
void wmma_gemm(const _Float16* __restrict__ A, int lda,
               const _Float16* __restrict__ B, int ldb,
               void* __restrict__ Cp, int ldc, int K) {
    int w    = threadIdx.x >> 5;
    int wm   = w & 1;              // 2 waves along M
    int wn   = w >> 1;             // 2 waves along N
    int m0   = blockIdx.y * 128 + wm * 64;
    int n0   = blockIdx.x * 128 + wn * 64;
    int lane = threadIdx.x & 31;
    int half = lane >> 4;
    int nlo  = lane & 15;

    // Per-lane fragment base pointers (hoisted; bumped by k inside loop)
    const _Float16* pa[4];
    const _Float16* pb[4];
#pragma unroll
    for (int i = 0; i < 4; ++i) {
        pa[i] = A + (size_t)(m0 + 16 * i + nlo) * lda + (half << 3);
        pb[i] = B + (size_t)(n0 + 16 * i + nlo) * ldb + (half << 3);
    }

    v8f acc[4][4];
#pragma unroll
    for (int i = 0; i < 4; ++i)
#pragma unroll
        for (int j = 0; j < 4; ++j) acc[i][j] = v8f{};

#pragma unroll 1
    for (int k0 = 0; k0 < K; k0 += 32) {
        v16h a[4], b[4];
#pragma unroll
        for (int i = 0; i < 4; ++i) a[i] = make_frag(pa[i] + k0);
#pragma unroll
        for (int i = 0; i < 4; ++i) b[i] = make_frag(pb[i] + k0);

        if (k0 + 32 < K) {  // uniform; pull next k-step toward L2/L1
            __builtin_prefetch(pa[0] + k0 + 32, 0, 1);
            __builtin_prefetch(pb[0] + k0 + 32, 0, 1);
        }

#pragma unroll
        for (int i = 0; i < 4; ++i)
#pragma unroll
            for (int j = 0; j < 4; ++j)
                acc[i][j] = __builtin_amdgcn_wmma_f32_16x16x32_f16(
                    false, a[i], false, b[j], (short)0, acc[i][j], false, false);
    }

    // C/D layout: lane L, acc element r -> (m = r + 8*(L>>4), n = L&15)
    if (F16OUT) {
        _Float16* C = (_Float16*)Cp;
#pragma unroll
        for (int i = 0; i < 4; ++i)
#pragma unroll
            for (int j = 0; j < 4; ++j)
#pragma unroll
                for (int r = 0; r < 8; ++r) {
                    int mr = m0 + 16 * i + r + half * 8;
                    C[(size_t)mr * ldc + n0 + 16 * j + nlo] = (_Float16)acc[i][j][r];
                }
    } else {
        float* C = (float*)Cp;
#pragma unroll
        for (int i = 0; i < 4; ++i)
#pragma unroll
            for (int j = 0; j < 4; ++j)
#pragma unroll
                for (int r = 0; r < 8; ++r) {
                    int mr = m0 + 16 * i + r + half * 8;
                    C[(size_t)mr * ldc + n0 + 16 * j + nlo] = acc[i][j][r];
                }
    }
}

// ---------------------------------------------------------------------------
// In-place RoPE on f16 (B*S rows, DIMD cols). Pair i: angle = s * 10000^(-2i/D)
// ---------------------------------------------------------------------------
__global__ void rope_kernel(_Float16* __restrict__ q, int total) {
    int idx = blockIdx.x * blockDim.x + threadIdx.x;
    if (idx >= total) return;
    int row = idx / (DIMD / 2);
    int i   = idx % (DIMD / 2);
    int s   = row % SEQ;
    // 10000^(-2i/D) = exp(-(2i/D) * ln(10000))
    float freq = __expf(-(2.0f * (float)i / (float)DIMD) * 9.210340371976184f);
    float ang  = (float)s * freq;
    float sn, cs;
    __sincosf(ang, &sn, &cs);
    _Float16* p = q + (size_t)row * DIMD + 2 * i;
    float xr = (float)p[0], xi = (float)p[1];
    p[0] = (_Float16)(xr * cs - xi * sn);
    p[1] = (_Float16)(xr * sn + xi * cs);
}

// ---------------------------------------------------------------------------
// Per-batch transpose: VT[b][d][s] = V[b][s][d]   (LDS 32x33 tile)
// grid = (SEQ/32, DIMD/32, BATCH), block = (32, 8)
// ---------------------------------------------------------------------------
__global__ void transpose_v(const _Float16* __restrict__ V, _Float16* __restrict__ VT) {
    __shared__ _Float16 tile[32][33];
    int b  = blockIdx.z;
    int s0 = blockIdx.x * 32;
    int d0 = blockIdx.y * 32;
    const _Float16* Vb = V  + (size_t)b * SEQ * DIMD;
    _Float16*      VTb = VT + (size_t)b * DIMD * SEQ;
    int tx = threadIdx.x, ty = threadIdx.y;
#pragma unroll
    for (int j = 0; j < 32; j += 8)
        tile[ty + j][tx] = Vb[(size_t)(s0 + ty + j) * DIMD + d0 + tx];
    __syncthreads();
#pragma unroll
    for (int j = 0; j < 32; j += 8)
        VTb[(size_t)(d0 + ty + j) * SEQ + s0 + tx] = tile[tx][ty + j];
}

// ---------------------------------------------------------------------------
// Row softmax with 1/sqrt(D) fold-in: P[row] = softmax(S[row] / 32) as f16.
// One 256-thread block per row of SEQ elements (each thread owns 8).
// ---------------------------------------------------------------------------
__global__ __launch_bounds__(256)
void softmax_row(const float* __restrict__ S, _Float16* __restrict__ P) {
    __shared__ float red[256];
    int row = blockIdx.x;
    int t   = threadIdx.x;
    const float* sr = S + (size_t)row * SEQ;
    _Float16*    pr = P + (size_t)row * SEQ;
    const float scale = 0.03125f;  // 1/sqrt(1024)

    float vals[8];
    float m = -1e30f;
#pragma unroll
    for (int j = 0; j < 8; ++j) {
        vals[j] = sr[t + j * 256] * scale;
        m = fmaxf(m, vals[j]);
    }
    red[t] = m;
    __syncthreads();
    for (int s = 128; s > 0; s >>= 1) {
        if (t < s) red[t] = fmaxf(red[t], red[t + s]);
        __syncthreads();
    }
    m = red[0];
    __syncthreads();

    float sum = 0.0f;
#pragma unroll
    for (int j = 0; j < 8; ++j) { vals[j] = __expf(vals[j] - m); sum += vals[j]; }
    red[t] = sum;
    __syncthreads();
    for (int s = 128; s > 0; s >>= 1) {
        if (t < s) red[t] += red[t + s];
        __syncthreads();
    }
    float inv = 1.0f / red[0];
#pragma unroll
    for (int j = 0; j < 8; ++j) pr[t + j * 256] = (_Float16)(vals[j] * inv);
}

// ---------------------------------------------------------------------------
// Host-side orchestration
// ---------------------------------------------------------------------------
extern "C" void kernel_launch(void* const* d_in, const int* in_sizes, int n_in,
                              void* d_out, int out_size, void* d_ws, size_t ws_size,
                              hipStream_t stream) {
    const float* x  = (const float*)d_in[0];
    const float* wq = (const float*)d_in[1];
    const float* wk = (const float*)d_in[2];
    const float* wv = (const float*)d_in[3];
    const float* wo = (const float*)d_in[4];
    float* out = (float*)d_out;

    const size_t ROWS = (size_t)BATCH * SEQ;          // 8192
    const size_t XSZ  = ROWS * DIMD;                  // 8M elems
    const size_t WSZ  = (size_t)DIMD * DIMD;          // 1M elems
    const size_t SSZ  = (size_t)SEQ * SEQ;            // 4M elems (per batch)

    // Workspace layout (bytes, 256-aligned)
    size_t off = 0;
    auto alloc = [&](size_t bytes) { size_t o = off; off += (bytes + 255) & ~(size_t)255; return o; };
    char* ws = (char*)d_ws;
    _Float16* Xh  = (_Float16*)(ws + alloc(XSZ * 2));
    _Float16* Wqh = (_Float16*)(ws + alloc(WSZ * 2));
    _Float16* Wkh = (_Float16*)(ws + alloc(WSZ * 2));
    _Float16* Wvh = (_Float16*)(ws + alloc(WSZ * 2));
    _Float16* Woh = (_Float16*)(ws + alloc(WSZ * 2));
    _Float16* Qh  = (_Float16*)(ws + alloc(XSZ * 2));
    _Float16* Kh  = (_Float16*)(ws + alloc(XSZ * 2));
    _Float16* Vh  = (_Float16*)(ws + alloc(XSZ * 2));
    _Float16* VTh = (_Float16*)(ws + alloc(XSZ * 2));
    float*    Sf  = (float*)   (ws + alloc(SSZ * 4)); // per-batch scores (reused)
    _Float16* Ph  = (_Float16*)(ws + alloc(SSZ * 2)); // per-batch probs  (reused)
    _Float16* AOh = (_Float16*)(ws + alloc(XSZ * 2));
    (void)ws_size;

    // 1) Convert inputs to f16
    cvt_f32_f16<<<(int)((XSZ + 255) / 256), 256, 0, stream>>>(x,  Xh,  (int)XSZ);
    cvt_f32_f16<<<(int)((WSZ + 255) / 256), 256, 0, stream>>>(wq, Wqh, (int)WSZ);
    cvt_f32_f16<<<(int)((WSZ + 255) / 256), 256, 0, stream>>>(wk, Wkh, (int)WSZ);
    cvt_f32_f16<<<(int)((WSZ + 255) / 256), 256, 0, stream>>>(wv, Wvh, (int)WSZ);
    cvt_f32_f16<<<(int)((WSZ + 255) / 256), 256, 0, stream>>>(wo, Woh, (int)WSZ);

    // 2) Projections: Q/K/V[8192x1024] = Xh * W^T   (grid: N/128 x M/128)
    {
        dim3 grid(DIMD / 128, (unsigned)(ROWS / 128));
        wmma_gemm<true><<<grid, 128, 0, stream>>>(Xh, DIMD, Wqh, DIMD, Qh, DIMD, DIMD);
        wmma_gemm<true><<<grid, 128, 0, stream>>>(Xh, DIMD, Wkh, DIMD, Kh, DIMD, DIMD);
        wmma_gemm<true><<<grid, 128, 0, stream>>>(Xh, DIMD, Wvh, DIMD, Vh, DIMD, DIMD);
    }

    // 3) RoPE on Q and K (in place)
    {
        int total  = (int)(ROWS * (DIMD / 2));
        int blocks = (total + 255) / 256;
        rope_kernel<<<blocks, 256, 0, stream>>>(Qh, total);
        rope_kernel<<<blocks, 256, 0, stream>>>(Kh, total);
    }

    // 4) Transpose V per batch: VT[b][d][s]
    {
        dim3 grid(SEQ / 32, DIMD / 32, BATCH);
        transpose_v<<<grid, dim3(32, 8), 0, stream>>>(Vh, VTh);
    }

    // 5) Attention per batch (stream-ordered reuse of Sf/Ph)
    for (int b = 0; b < BATCH; ++b) {
        const _Float16* Qb  = Qh  + (size_t)b * SEQ * DIMD;
        const _Float16* Kb  = Kh  + (size_t)b * SEQ * DIMD;
        const _Float16* VTb = VTh + (size_t)b * DIMD * SEQ;
        _Float16*       AOb = AOh + (size_t)b * SEQ * DIMD;

        // scores[2048x2048] = Qb * Kb^T  (K-dim 1024), f32 out
        wmma_gemm<false><<<dim3(SEQ / 128, SEQ / 128), 128, 0, stream>>>(
            Qb, DIMD, Kb, DIMD, Sf, SEQ, DIMD);

        // P = softmax(scores / 32), f16
        softmax_row<<<SEQ, 256, 0, stream>>>(Sf, Ph);

        // AO[2048x1024] = P * VT^T  (K-dim 2048), f16 out
        wmma_gemm<true><<<dim3(DIMD / 128, SEQ / 128), 128, 0, stream>>>(
            Ph, SEQ, VTb, SEQ, AOb, DIMD, SEQ);
    }

    // 6) Output projection: out[8192x1024] = AOh * wo^T, f32 out
    wmma_gemm<false><<<dim3(DIMD / 128, (unsigned)(ROWS / 128)), 128, 0, stream>>>(
        AOh, DIMD, Woh, DIMD, out, DIMD, DIMD);
}
